// MoEAudioProjector_34548716929040
// MI455X (gfx1250) — matrix operand order repaired and checked by reading, not compiled, not run on hardware
//
#include <hip/hip_runtime.h>
#include <hip/hip_bf16.h>

// ---------------- problem constants ----------------
#define BATCH     4
#define SEQ_S     1024
#define DIM_D     1280
#define FOLD_K    4
#define DIM_IN    5120        // D*K
#define DIM_H     2048
#define DIM_O     2048
#define NEXP      4
#define NSLOT     5           // 4 experts + shared (slot 4)
#define TK        1024        // B * S/K tokens

// ---------------- GEMM tiling ----------------
#define BM   128
#define BN   128
#define BK   64
#define LDK  72               // padded LDS row (bf16 elems) = 144B; TDM pads 16B/row
#define TILE_BYTES (BM * LDK * 2)   // 18432

typedef __attribute__((ext_vector_type(16))) __bf16 bf16x16;
typedef __attribute__((ext_vector_type(8)))  float  floatx8;
typedef __attribute__((ext_vector_type(4)))  unsigned int uint32x4;
typedef __attribute__((ext_vector_type(8)))  int          int32x8;
typedef __attribute__((ext_vector_type(4)))  int          int32x4;
typedef __attribute__((ext_vector_type(4)))  unsigned int u32x4;

// LDS (addrspace 3) pointers built from 32-bit LDS byte offsets -> ds_load/ds_store
typedef const __attribute__((address_space(3))) u32x4* lds_cu4p;
typedef       __attribute__((address_space(3))) float* lds_f32p;

union FragB16 { bf16x16 v; u32x4 q[2]; };

__device__ __forceinline__ unsigned short f2bf(float f) {
  unsigned int u = __float_as_uint(f);
  if ((u & 0x7fffffffu) > 0x7f800000u) return (unsigned short)((u >> 16) | 0x40u); // NaN
  return (unsigned short)((u + 0x7fffu + ((u >> 16) & 1u)) >> 16);                 // RNE
}

// ---------------------------------------------------------------------------
// Issue one TDM 2D tile load: 128 rows x 64 bf16 from a row-major [rows][Kdim]
// matrix into LDS at lds_off, padding 16B after every 128B row (-> LDK=72).
// Wave-uniform; EXEC ignored by TDM; tracked on the issuing wave's TENSORcnt.
// ---------------------------------------------------------------------------
__device__ __forceinline__ void tdm_issue_tile(const unsigned short* gptr,
                                               unsigned int lds_off, int Kdim) {
  unsigned long long ga = (unsigned long long)(size_t)gptr;
  uint32x4 g0;
  g0.x = 1u;                                                   // count=1, user mode
  g0.y = lds_off;                                              // lds_addr (bytes)
  g0.z = (unsigned int)(ga & 0xffffffffu);                     // global_addr[31:0]
  g0.w = (unsigned int)((ga >> 32) & 0x01ffffffu) | (2u << 30); // addr[56:32]|type=2
  int32x8 g1;
  g1[0] = (1 << 16)     // data_size = 2 bytes
        | (1 << 20)     // pad_enable
        | (4 << 22)     // pad_interval: 2^(4+1)=32 DWORDs = 128B
        | (3 << 25);    // pad_amount: 4 DWORDs = 16B
  g1[1] = (Kdim & 0xffff) << 16;                 // tensor_dim0[15:0]
  g1[2] = ((Kdim >> 16) & 0xffff);               // tensor_dim0[31:16] | dim1 lo = 0
  g1[3] = 0x0010 | (BK << 16);                   // tensor_dim1 = 1<<20; tile_dim0=64
  g1[4] = BM;                                    // tile_dim1 = 128; tile_dim2 = 0
  g1[5] = Kdim;                                  // tensor_dim0_stride[31:0]
  g1[6] = 0;
  g1[7] = 0;
  int32x4 z4 = {0, 0, 0, 0};
  int32x8 z8 = {0, 0, 0, 0, 0, 0, 0, 0};
  __builtin_amdgcn_tensor_load_to_lds(g0, g1, z4, z4, z8, 0);
}

// ---------------------------------------------------------------------------
// WMMA mainloop with TDM double-buffering: C[128x128] += A[128xK] * W[128xK]^T.
// Wave 0 issues next-stage DMA, waits tensorcnt<=2, all waves barrier+compute.
// All LDS reads via addrspace(3) 32-bit offsets -> ds_load_b128.
// ---------------------------------------------------------------------------
__device__ __forceinline__ void gemm_tile_mainloop(
    const unsigned short* __restrict__ A,     // [M][Kdim]
    const unsigned short* __restrict__ W,     // [N][Kdim]
    int Kdim, int mBase, int nBase,
    int tid, int wm, int wn, int lm, int lhalf,
    unsigned int smemLds,
    floatx8 acc[4][2])
{
  const int wave = tid >> 5;
  const int nS = Kdim / BK;
  // per-lane base LDS byte offsets of the A / B fragments within a tile buffer
  const unsigned int aFragBase = (unsigned int)(((wm * 64 + lm) * LDK + lhalf * 8) * 2);
  const unsigned int bFragBase = (unsigned int)(((wn * 32 + lm) * LDK + lhalf * 8) * 2);

  if (wave == 0) {  // prologue: stage 0 into buffer 0
    tdm_issue_tile(A + (size_t)mBase * Kdim, smemLds, Kdim);
    tdm_issue_tile(W + (size_t)nBase * Kdim, smemLds + TILE_BYTES, Kdim);
  }

  for (int s = 0; s < nS; ++s) {
    const unsigned int bufOff = smemLds + (unsigned int)(s & 1) * (2 * TILE_BYTES);
    const unsigned int nxtOff = smemLds + (unsigned int)(~s & 1) * (2 * TILE_BYTES);
    if (wave == 0) {
      if (s + 1 < nS) {  // overlap next stage DMA with this stage's WMMAs
        tdm_issue_tile(A + (size_t)mBase * Kdim + (size_t)(s + 1) * BK, nxtOff, Kdim);
        tdm_issue_tile(W + (size_t)nBase * Kdim + (size_t)(s + 1) * BK,
                       nxtOff + TILE_BYTES, Kdim);
        __builtin_amdgcn_s_wait_tensorcnt(2);   // stage s landed; s+1 in flight
      } else {
        __builtin_amdgcn_s_wait_tensorcnt(0);
      }
    }
    __syncthreads();    // publish buf[cur]

#pragma unroll
    for (int ks = 0; ks < BK; ks += 32) {
      FragB16 af[4];
#pragma unroll
      for (int i = 0; i < 4; ++i) {
        // A 16x32 bf16 frag: lane(0..15)=row M, halves hold K {h*8..+8},{16+h*8..+8}
        unsigned int p = bufOff + aFragBase + (unsigned int)((i * 16 * LDK + ks) * 2);
        af[i].q[0] = *(lds_cu4p)p;
        af[i].q[1] = *(lds_cu4p)(p + 32);
      }
      FragB16 bfr[2];
#pragma unroll
      for (int j = 0; j < 2; ++j) {
        // B 32x16 bf16 frag: lane(0..15)=col N, halves hold K {h*8..+8},{16+h*8..+8}
        unsigned int p = bufOff + TILE_BYTES + bFragBase
                       + (unsigned int)((j * 16 * LDK + ks) * 2);
        bfr[j].q[0] = *(lds_cu4p)p;
        bfr[j].q[1] = *(lds_cu4p)(p + 32);
      }
#pragma unroll
      for (int i = 0; i < 4; ++i)
#pragma unroll
        for (int j = 0; j < 2; ++j)
          acc[i][j] = __builtin_amdgcn_wmma_f32_16x16x32_bf16(
              false, af[i].v, false, bfr[j].v, (short)0, acc[i][j], false, false);
    }
    __syncthreads();    // all reads of buf[cur] done before it is refilled
  }
}

// ---------------------------------------------------------------------------
// Kernel 1: depthwise conv3 + residual + fold(K=4) + RMSNorm + sigmoid router
// ---------------------------------------------------------------------------
__global__ __launch_bounds__(256) void k_conv_norm_router(
    const float* __restrict__ x,        // [B,S,D]
    const float* __restrict__ conv_w,   // [D,1,3]
    const float* __restrict__ conv_b,   // [D]
    const float* __restrict__ rms_w,    // [IN]
    const float* __restrict__ router_w, // [E,IN]
    unsigned short* __restrict__ nbuf,  // [TK,IN] bf16
    float* __restrict__ combine)        // [TK,E]
{
  __shared__ float red[256];
  __shared__ float lg[4];
  int t   = blockIdx.x;
  int b   = t >> 8;
  int s0  = (t & 255) << 2;
  int tid = threadIdx.x;

  float v[20];
  float ss = 0.f;
#pragma unroll
  for (int j = 0; j < 20; ++j) {
    int i    = tid + 256 * j;
    int srel = i / DIM_D;
    int d    = i - srel * DIM_D;
    int s    = s0 + srel;
    size_t base = ((size_t)b * SEQ_S + s) * DIM_D + d;
    float xc = x[base];
    float xm = (s > 0)         ? x[base - DIM_D] : 0.f;
    float xp = (s < SEQ_S - 1) ? x[base + DIM_D] : 0.f;
    float y = xc + conv_w[d * 3 + 0] * xm + conv_w[d * 3 + 1] * xc
                 + conv_w[d * 3 + 2] * xp + conv_b[d];
    v[j] = y;
    ss  += y * y;
  }
  red[tid] = ss;
  __syncthreads();
  for (int off = 128; off > 0; off >>= 1) {
    if (tid < off) red[tid] += red[tid + off];
    __syncthreads();
  }
  float inv = rsqrtf(red[0] * (1.f / DIM_IN) + 1e-8f);

  float dots[4] = {0.f, 0.f, 0.f, 0.f};
#pragma unroll
  for (int j = 0; j < 20; ++j) {
    int i    = tid + 256 * j;
    float nv = v[j] * inv * rms_w[i];
    nbuf[(size_t)t * DIM_IN + i] = f2bf(nv);
#pragma unroll
    for (int e = 0; e < 4; ++e) dots[e] += nv * router_w[e * DIM_IN + i];
  }
  for (int e = 0; e < 4; ++e) {
    __syncthreads();
    red[tid] = dots[e];
    __syncthreads();
    for (int off = 128; off > 0; off >>= 1) {
      if (tid < off) red[tid] += red[tid + off];
      __syncthreads();
    }
    if (tid == 0) lg[e] = red[0];
  }
  __syncthreads();
  if (tid == 0) {
    float p[4];
    for (int e = 0; e < 4; ++e) p[e] = 1.f / (1.f + expf(-lg[e]));
    int i0 = 0;
    for (int e = 1; e < 4; ++e) if (p[e] > p[i0]) i0 = e;      // ties -> lowest idx
    int i1 = (i0 == 0) ? 1 : 0;
    for (int e = 0; e < 4; ++e) if (e != i0 && p[e] > p[i1]) i1 = e;
    float denom = p[i0] + p[i1] + 1e-6f;
    float c[4] = {0.f, 0.f, 0.f, 0.f};
    c[i0] = p[i0] / denom;
    c[i1] = p[i1] / denom;
    for (int e = 0; e < 4; ++e) combine[t * 4 + e] = c[e];
  }
}

// ---------------------------------------------------------------------------
// Kernel 2: grid-stride f32 -> bf16 (packed float4 -> 4x bf16)
// ---------------------------------------------------------------------------
__global__ void k_f32_to_bf16(const float* __restrict__ in,
                              unsigned short* __restrict__ out, long n4) {
  long stride = (long)gridDim.x * blockDim.x;
  for (long i = (long)blockIdx.x * blockDim.x + threadIdx.x; i < n4; i += stride) {
    float4 f = ((const float4*)in)[i];
    ushort4 o;
    o.x = f2bf(f.x); o.y = f2bf(f.y); o.z = f2bf(f.z); o.w = f2bf(f.w);
    ((ushort4*)out)[i] = o;
  }
}

// ---------------------------------------------------------------------------
// Kernel 3: layer-1 GEMM  hidden[e] = ReLU(n @ W1[e]^T + b1[e])  -> bf16
// grid (H/BN=16, TK/BM=8, NSLOT=5), dynamic LDS = 4*TILE_BYTES
// ---------------------------------------------------------------------------
__global__ __launch_bounds__(256) void k_gemm1(
    const unsigned short* __restrict__ nbuf,  // [TK][IN]
    const unsigned short* __restrict__ w1,    // [5][H][IN]
    const float* __restrict__ eb1,            // [4][H]
    const float* __restrict__ sb1,            // [H]
    unsigned short* __restrict__ hid)         // [5][TK][H]
{
  extern __shared__ char smem[];
  unsigned int smemLds = (unsigned int)(size_t)(void*)smem;  // LDS byte offset
  int e     = blockIdx.z;
  int mBase = blockIdx.y * BM;
  int nBase = blockIdx.x * BN;
  int tid   = threadIdx.x;
  int wave  = tid >> 5, lane = tid & 31;
  int wm    = wave >> 2, wn = wave & 3;
  int lm    = lane & 15, lhalf = lane >> 4;

  floatx8 acc[4][2];
#pragma unroll
  for (int i = 0; i < 4; ++i)
#pragma unroll
    for (int j = 0; j < 2; ++j)
#pragma unroll
      for (int r = 0; r < 8; ++r) acc[i][j][r] = 0.f;

  const unsigned short* W = w1 + (size_t)e * DIM_H * DIM_IN;
  gemm_tile_mainloop(nbuf, W, DIM_IN, mBase, nBase, tid, wm, wn, lm, lhalf,
                     smemLds, acc);

#pragma unroll
  for (int i = 0; i < 4; ++i) {
    int rowBase = mBase + wm * 64 + i * 16 + lhalf * 8;
#pragma unroll
    for (int j = 0; j < 2; ++j) {
      int col    = nBase + wn * 32 + j * 16 + lm;
      float bias = (e < 4) ? eb1[e * DIM_H + col] : sb1[col];
#pragma unroll
      for (int r = 0; r < 8; ++r) {
        float v = acc[i][j][r] + bias;
        v = v > 0.f ? v : 0.f;
        hid[((size_t)e * TK + rowBase + r) * DIM_H + col] = f2bf(v);
      }
    }
  }
}

// ---------------------------------------------------------------------------
// Kernel 4: layer-2 GEMM over all 5 slots with combine weighting.
// out[t,o] = sum_e w_e(t) * (hidden[e] @ W2[e]^T + b2[e]),  w_shared = 1.
// grid (O/BN=16, TK/BM=8), dynamic LDS = 4*TILE_BYTES + BM*4*4
// ---------------------------------------------------------------------------
__global__ __launch_bounds__(256) void k_gemm2(
    const unsigned short* __restrict__ hid,   // [5][TK][H]
    const unsigned short* __restrict__ w2,    // [5][O][H]
    const float* __restrict__ eb2,            // [4][O]
    const float* __restrict__ sb2,            // [O]
    const float* __restrict__ combine,        // [TK][4]
    float* __restrict__ out)                  // [TK][O]
{
  extern __shared__ char smem[];
  unsigned int smemLds = (unsigned int)(size_t)(void*)smem;
  unsigned int combOff = smemLds + 4 * TILE_BYTES;           // [BM][4] f32 in LDS
  int mBase = blockIdx.y * BM;
  int nBase = blockIdx.x * BN;
  int tid   = threadIdx.x;
  int wave  = tid >> 5, lane = tid & 31;
  int wm    = wave >> 2, wn = wave & 3;
  int lm    = lane & 15, lhalf = lane >> 4;

  for (int i = tid; i < BM * 4; i += 256)
    ((lds_f32p)combOff)[i] = combine[(size_t)mBase * 4 + i];

  floatx8 tot[4][2];
#pragma unroll
  for (int i = 0; i < 4; ++i)
#pragma unroll
    for (int j = 0; j < 2; ++j)
#pragma unroll
      for (int r = 0; r < 8; ++r) tot[i][j][r] = 0.f;

  for (int e = 0; e < NSLOT; ++e) {
    floatx8 acc[4][2];
#pragma unroll
    for (int i = 0; i < 4; ++i)
#pragma unroll
      for (int j = 0; j < 2; ++j)
#pragma unroll
        for (int r = 0; r < 8; ++r) acc[i][j][r] = 0.f;

    const unsigned short* A = hid + (size_t)e * TK * DIM_H;
    const unsigned short* W = w2 + (size_t)e * DIM_O * DIM_H;
    gemm_tile_mainloop(A, W, DIM_H, mBase, nBase, tid, wm, wn, lm, lhalf,
                       smemLds, acc);

#pragma unroll
    for (int i = 0; i < 4; ++i) {
      int rlBase = wm * 64 + i * 16 + lhalf * 8;
#pragma unroll
      for (int j = 0; j < 2; ++j) {
        int col    = nBase + wn * 32 + j * 16 + lm;
        float bias = (e < 4) ? eb2[e * DIM_O + col] : sb2[col];
#pragma unroll
        for (int r = 0; r < 8; ++r) {
          float w = (e == 4) ? 1.f : ((lds_f32p)combOff)[(rlBase + r) * 4 + e];
          tot[i][j][r] += w * (acc[i][j][r] + bias);
        }
      }
    }
  }

#pragma unroll
  for (int i = 0; i < 4; ++i) {
    int rowBase = mBase + wm * 64 + i * 16 + lhalf * 8;
#pragma unroll
    for (int j = 0; j < 2; ++j) {
      int col = nBase + wn * 32 + j * 16 + lm;
#pragma unroll
      for (int r = 0; r < 8; ++r)
        out[(size_t)(rowBase + r) * DIM_O + col] = tot[i][j][r];
    }
  }
}

// ---------------------------------------------------------------------------
extern "C" void kernel_launch(void* const* d_in, const int* in_sizes, int n_in,
                              void* d_out, int out_size, void* d_ws, size_t ws_size,
                              hipStream_t stream) {
  (void)in_sizes; (void)n_in; (void)out_size; (void)ws_size;
  const float* x        = (const float*)d_in[0];
  const float* conv_w   = (const float*)d_in[1];
  const float* conv_b   = (const float*)d_in[2];
  const float* rms_w    = (const float*)d_in[3];
  const float* router_w = (const float*)d_in[4];
  const float* sw1      = (const float*)d_in[5];
  const float* sb1      = (const float*)d_in[6];
  const float* sw2      = (const float*)d_in[7];
  const float* sb2      = (const float*)d_in[8];
  const float* ew1      = (const float*)d_in[9];
  const float* eb1      = (const float*)d_in[10];
  const float* ew2      = (const float*)d_in[11];
  const float* eb2      = (const float*)d_in[12];
  float* out = (float*)d_out;

  char* ws = (char*)d_ws;
  size_t off = 0;
  auto take = [&](size_t bytes) {
    void* p = ws + off;
    off += (bytes + 255) & ~(size_t)255;
    return p;
  };
  unsigned short* nbuf = (unsigned short*)take((size_t)TK * DIM_IN * 2);
  unsigned short* w1b  = (unsigned short*)take((size_t)NSLOT * DIM_H * DIM_IN * 2);
  unsigned short* w2b  = (unsigned short*)take((size_t)NSLOT * DIM_O * DIM_H * 2);
  unsigned short* hidb = (unsigned short*)take((size_t)NSLOT * TK * DIM_H * 2);
  float* comb          = (float*)take((size_t)TK * NEXP * 4);

  // 1) conv + fold + RMSNorm + router
  k_conv_norm_router<<<TK, 256, 0, stream>>>(x, conv_w, conv_b, rms_w, router_w,
                                             nbuf, comb);

  // 2) weight conversion f32 -> bf16 (experts in slots 0..3, shared in slot 4)
  const int CB = 2048;
  k_f32_to_bf16<<<CB, 256, 0, stream>>>(ew1, w1b,
      (long)NEXP * DIM_H * DIM_IN / 4);
  k_f32_to_bf16<<<CB, 256, 0, stream>>>(sw1, w1b + (size_t)NEXP * DIM_H * DIM_IN,
      (long)DIM_H * DIM_IN / 4);
  k_f32_to_bf16<<<CB, 256, 0, stream>>>(ew2, w2b,
      (long)NEXP * DIM_O * DIM_H / 4);
  k_f32_to_bf16<<<CB, 256, 0, stream>>>(sw2, w2b + (size_t)NEXP * DIM_O * DIM_H,
      (long)DIM_O * DIM_H / 4);

  // 3) layer-1 GEMMs for 5 slots (TDM-fed, double buffered)
  dim3 g1(DIM_H / BN, TK / BM, NSLOT);
  k_gemm1<<<g1, 256, 4 * TILE_BYTES, stream>>>(nbuf, w1b, eb1, sb1, hidb);

  // 4) layer-2 GEMMs + combine weighting + final store
  dim3 g2(DIM_O / BN, TK / BM, 1);
  k_gemm2<<<g2, 256, 4 * TILE_BYTES + BM * 4 * 4, stream>>>(hidb, w2b, eb2, sb2,
                                                            comb, out);
}
